// MultiHeadAttention_8881992368783
// MI455X (gfx1250) — compile-verified, matrix operand order checked by reading, not run on hardware
//
#include <hip/hip_runtime.h>
#include <stdint.h>

#define HD __device__ __forceinline__

typedef __bf16 bh;
typedef __attribute__((ext_vector_type(16))) __bf16 v16bf;
typedef __attribute__((ext_vector_type(8)))  float  v8f;
typedef __attribute__((ext_vector_type(4)))  unsigned int u32x4;
typedef __attribute__((ext_vector_type(8)))  int         i32x8;
typedef __attribute__((ext_vector_type(4)))  int         i32x4;

static constexpr int Bsz = 2, Sseq = 2048, Dmodel = 1024, Hh = 16, Dk = 64;
static constexpr int Mrows = Bsz * Sseq; // 4096

#if defined(__AMDGCN__) && defined(__has_builtin)
#if __has_builtin(__builtin_amdgcn_tensor_load_to_lds) && \
    __has_builtin(__builtin_amdgcn_s_wait_tensorcnt)
#define HAVE_TDM 1
#endif
#endif
#ifndef HAVE_TDM
#define HAVE_TDM 0
#endif

// ---- fp32 -> bf16, round-to-nearest-even ----
HD bh f2bf(float f) {
  union { float f; uint32_t u; } x; x.f = f;
  uint32_t r = x.u + 0x7FFFu + ((x.u >> 16) & 1u);
  unsigned short h = (unsigned short)(r >> 16);
  return __builtin_bit_cast(__bf16, h);
}

// Build a 16-bit A/B WMMA fragment from LDS laid out so the contraction dim
// is contiguous. Per ISA 16-bit K-map: elems 0..7 -> K = half*8 + 0..7,
// elems 8..15 -> K = 16 + half*8 + 0..7.  Two ds_load_b128 per fragment.
union Frag { v16bf v; uint4 q[2]; };
HD v16bf load_frag(const bh* base, int hf) {
  Frag u;
  u.q[0] = *(const uint4*)(base + hf * 8);
  u.q[1] = *(const uint4*)(base + 16 + hf * 8);
  return u.v;
}

#if HAVE_TDM
// Issue a TDM 2D tile load: 64 rows x 64 bf16 cols from a row-major
// [tensorRows x 64] bf16 tensor into LDS at lds_off, with LDS padding of
// 4 dwords after every 32 dwords (row stride 128B -> 144B = 72 elements).
HD void tdm_load_64x64(uint32_t lds_off, const bh* gtile, uint32_t tensorRows) {
  uint64_t ga = (uint64_t)(uintptr_t)gtile;
  u32x4 g0;
  g0[0] = 1u;                                   // count=1, user descriptor
  g0[1] = lds_off;                              // lds_addr (bytes)
  g0[2] = (uint32_t)ga;                         // global_addr[31:0]
  g0[3] = (uint32_t)((ga >> 32) & 0x01FFFFFFu)  // global_addr[56:32]
          | (2u << 30);                         // type = 2 ("image")
  i32x8 g1;
  // word0: wg_mask=0 | data_size=1 (2B) | pad_enable | pad_interval=4 (32dw)
  //        | pad_amount=3 (4 dwords)
  g1[0] = (int)((1u << 16) | (1u << 20) | (4u << 22) | (3u << 25));
  g1[1] = (int)((64u & 0xFFFFu) << 16);         // tensor_dim0[15:0] = 64
  g1[2] = (int)((tensorRows & 0xFFFFu) << 16);  // dim0 hi=0 | tensor_dim1 lo
  g1[3] = (int)((tensorRows >> 16) & 0xFFFFu) | (int)(64u << 16); // tile_dim0=64
  g1[4] = (int)(64u);                           // tile_dim1=64, tile_dim2=0
  g1[5] = (int)(64u);                           // tensor_dim0_stride = 64
  g1[6] = 0;
  g1[7] = 0;
  i32x4 z4 = {};
#if __clang_major__ >= 23
  i32x8 z8 = {};
  __builtin_amdgcn_tensor_load_to_lds(g0, g1, z4, z4, z8, 0);
#else
  __builtin_amdgcn_tensor_load_to_lds(g0, g1, z4, z4, 0);
#endif
}
#endif

// ---------------- tiled WMMA GEMM: C[M,N] = A[M,K] * W[K,N] ----------------
// Block = 256 threads (8 waves). Tile: 128(M) x 64(N), K-step 32.
// Double-buffered LDS pipeline, one barrier per K-step: global loads for
// tile i+1 are issued before the barrier so VMEM latency hides under the
// barrier wait + WMMA compute of tile i.
// TA = float: convert fp32->bf16 while staging A; TA = bh: direct bf16.
// W is always fp32, converted + transposed while staging.
// MODE 0: store bf16 head-major [B*H][S][64]   (projection path)
// MODE 1: store fp32 row-major  [M][1024]      (final output path)
template<int MODE, typename TA>
__global__ __launch_bounds__(256) void k_gemm(const TA* __restrict__ A,
                                              const float* __restrict__ W,
                                              void* __restrict__ out,
                                              int Kdim, int Ncols) {
  __shared__ __align__(16) bh As[2][128 * 40];  // [row][k], stride 40
  __shared__ __align__(16) bh Bs[2][64 * 40];   // [n][k]  (transposed)
  const int tid  = threadIdx.x;
  const int w    = tid >> 5, lane = tid & 31;
  const int ln   = lane & 15, hf = lane >> 4;
  const int rowBase = blockIdx.y * 128, nBase = blockIdx.x * 64;

  float4 fa[4];  // staged A (fp32 path)
  uint4  ua[2];  // staged A (bf16 path)
  float4 fw[2];  // staged W

  auto g_load = [&](int k0) {
    if constexpr (__is_same(TA, float)) {
#pragma unroll
      for (int i = 0; i < 4; ++i) {
        int idx = tid + i * 256;          // 1024 float4 segments
        int r = idx >> 3, cs = idx & 7;
        fa[i] = *(const float4*)&A[(size_t)(rowBase + r) * Kdim + k0 + cs * 4];
      }
    } else {
#pragma unroll
      for (int i = 0; i < 2; ++i) {
        int idx = tid + i * 256;          // 512 uint4 segments
        int r = idx >> 2, cs = idx & 3;
        ua[i] = *(const uint4*)&A[(size_t)(rowBase + r) * Kdim + k0 + cs * 8];
      }
    }
#pragma unroll
    for (int i = 0; i < 2; ++i) {
      int idx = tid + i * 256;            // 512 float4 segments
      int kr = idx >> 4, ns = idx & 15;
      fw[i] = *(const float4*)&W[(size_t)(k0 + kr) * Ncols + nBase + ns * 4];
    }
  };
  auto s_store = [&](int buf) {
    if constexpr (__is_same(TA, float)) {
#pragma unroll
      for (int i = 0; i < 4; ++i) {
        int idx = tid + i * 256;
        int r = idx >> 3, cs = idx & 7;
        bh t4[4] = {f2bf(fa[i].x), f2bf(fa[i].y), f2bf(fa[i].z), f2bf(fa[i].w)};
        *(uint2*)&As[buf][r * 40 + cs * 4] = *(uint2*)t4;
      }
    } else {
#pragma unroll
      for (int i = 0; i < 2; ++i) {
        int idx = tid + i * 256;
        int r = idx >> 2, cs = idx & 3;
        *(uint4*)&As[buf][r * 40 + cs * 8] = ua[i];
      }
    }
#pragma unroll
    for (int i = 0; i < 2; ++i) {
      int idx = tid + i * 256;
      int kr = idx >> 4, ns = idx & 15;
      Bs[buf][(ns * 4 + 0) * 40 + kr] = f2bf(fw[i].x);
      Bs[buf][(ns * 4 + 1) * 40 + kr] = f2bf(fw[i].y);
      Bs[buf][(ns * 4 + 2) * 40 + kr] = f2bf(fw[i].z);
      Bs[buf][(ns * 4 + 3) * 40 + kr] = f2bf(fw[i].w);
    }
  };

  v8f acc[4] = {};
  auto compute = [&](int buf) {
    v16bf af = load_frag(&As[buf][(w * 16 + ln) * 40], hf);
#pragma unroll
    for (int t = 0; t < 4; ++t) {
      v16bf bfg = load_frag(&Bs[buf][(t * 16 + ln) * 40], hf);
      acc[t] = __builtin_amdgcn_wmma_f32_16x16x32_bf16(
          false, af, false, bfg, (short)0, acc[t], false, false);
    }
  };

  g_load(0);
  s_store(0);
  const int T = Kdim / 32;
  for (int i = 0; i < T; ++i) {
    if (i + 1 < T) g_load((i + 1) * 32);  // prefetch next tile into regs
    __syncthreads();                      // stores of buf (i&1) visible
    compute(i & 1);
    if (i + 1 < T) s_store((i + 1) & 1);  // opposite buffer: no conflict
  }

  // D layout: lane holds col n = ln; VGPR r holds row m = r + 8*hf
#pragma unroll
  for (int t = 0; t < 4; ++t)
#pragma unroll
    for (int r = 0; r < 8; ++r) {
      int m   = r + 8 * hf;
      int row = rowBase + w * 16 + m;
      int col = nBase + t * 16 + ln;
      if (MODE == 0) {
        int b = row >> 11, s = row & 2047, h = col >> 6, d = col & 63;
        ((bh*)out)[((((size_t)(b * Hh + h)) << 11) + s) * 64 + d] =
            f2bf(acc[t][r]);
      } else {
        ((float*)out)[(size_t)row * 1024 + col] = acc[t][r];
      }
    }
}

// ---------------- causal flash attention ----------------
// Grid: (S/64, B*H). Block = 128 threads (4 waves), each wave owns 16 q-rows.
// K tiles are TDM-prefetched one iteration ahead into a ping-pong buffer.
static constexpr uint32_t TILE_SZ = 64 * 72 * 2;       // 9216 B
static constexpr uint32_t QS_OFF  = 0;
static constexpr uint32_t KS_OFF  = TILE_SZ;           // 2 buffers
static constexpr uint32_t VT_OFF  = 3 * TILE_SZ;
static constexpr uint32_t PS_OFF  = 4 * TILE_SZ;
static constexpr uint32_t FLASH_SMEM = 5 * TILE_SZ;    // 46080 B

__global__ __launch_bounds__(128) void k_flash(const bh* __restrict__ qh,
                                               const bh* __restrict__ kh,
                                               const bh* __restrict__ vh,
                                               bh* __restrict__ attn) {
  extern __shared__ __align__(16) char smem[];
  bh* Qs = (bh*)(smem + QS_OFF);       // [qrow][d],  stride 72
  bh* Vt = (bh*)(smem + VT_OFF);       // [dv][key],  stride 72 (transposed)
  bh* Ps = (bh*)(smem + PS_OFF);       // per-wave P scratch [m][key]
  const int tid  = threadIdx.x;
  const int w    = tid >> 5, lane = tid & 31;
  const int ln   = lane & 15, hf = lane >> 4;
  const int qt   = blockIdx.x;
  const int bhid = blockIdx.y;               // b*H + h
  const int bq = bhid / Hh, h = bhid % Hh;
  const size_t headBase = (size_t)bhid * Sseq * Dk;
  const int qBase = qt * 64;

  // ---- prologue: stage Q tile and K tile 0 ----
#if HAVE_TDM
  if (w == 0) {
    tdm_load_64x64(QS_OFF, qh + headBase + (size_t)qBase * 64, Sseq);
    tdm_load_64x64(KS_OFF, kh + headBase, Sseq);  // K tile kt=0
  }
#else
#pragma unroll
  for (int i = 0; i < 4; ++i) {
    int idx = tid + i * 128;
    int r = idx >> 3, cs = idx & 7;
    *(uint4*)&Qs[r * 72 + cs * 8] =
        *(const uint4*)&qh[headBase + (size_t)(qBase + r) * 64 + cs * 8];
  }
#endif

  v8f co[4] = {};
  float mrow[8], lrow[8];
#pragma unroll
  for (int r = 0; r < 8; ++r) { mrow[r] = -1e30f; lrow[r] = 0.f; }
  bh* Pw = &Ps[w * 16 * 72];

  for (int kt = 0; kt <= qt; ++kt) {
    __syncthreads();  // all waves done with tile kt-1 buffers
    const int kBase = kt * 64;
#if HAVE_TDM
    // prefetch K tile kt+1 into the opposite buffer (safe after the barrier:
    // buffer (kt+1)&1 was last read during iteration kt-1)
    if (w == 0 && kt < qt)
      tdm_load_64x64(KS_OFF + ((uint32_t)((kt + 1) & 1)) * TILE_SZ,
                     kh + headBase + (size_t)(kBase + 64) * 64, Sseq);
#else
    {
      bh* Ksb = (bh*)(smem + KS_OFF + (uint32_t)(kt & 1) * TILE_SZ);
#pragma unroll
      for (int i = 0; i < 4; ++i) {
        int idx = tid + i * 128;
        int r = idx >> 3, cs = idx & 7;
        *(uint4*)&Ksb[r * 72 + cs * 8] =
            *(const uint4*)&kh[headBase + (size_t)(kBase + r) * 64 + cs * 8];
      }
    }
#endif
    // ---- stage V tile transposed (manual: TDM has no transpose) ----
#pragma unroll
    for (int i = 0; i < 4; ++i) {
      int idx = tid + i * 128;
      int key = idx >> 3, ds = idx & 7;
      uint4 vq = *(const uint4*)&vh[headBase + (size_t)(kBase + key) * 64 + ds * 8];
      bh tmp[8]; *(uint4*)tmp = vq;
#pragma unroll
      for (int j = 0; j < 8; ++j) Vt[(ds * 8 + j) * 72 + key] = tmp[j];
    }
#if HAVE_TDM
    // TDM completes in order: <=1 outstanding means tile kt (and Q) landed.
    if (w == 0) {
      if (kt < qt) __builtin_amdgcn_s_wait_tensorcnt(1);
      else         __builtin_amdgcn_s_wait_tensorcnt(0);
    }
#endif
    __syncthreads();
    const bh* Ksb = (const bh*)(smem + KS_OFF + (uint32_t)(kt & 1) * TILE_SZ);

    // S = Q Kt : 16x64 scores per wave (contraction over d=64, two K=32 steps)
    v8f sacc[4] = {};
#pragma unroll
    for (int kk = 0; kk < 64; kk += 32) {
      v16bf af = load_frag(&Qs[(w * 16 + ln) * 72 + kk], hf);
#pragma unroll
      for (int t = 0; t < 4; ++t) {
        v16bf bfg = load_frag(&Ksb[(t * 16 + ln) * 72 + kk], hf);
        sacc[t] = __builtin_amdgcn_wmma_f32_16x16x32_bf16(
            false, af, false, bfg, (short)0, sacc[t], false, false);
      }
    }
    // scale + causal mask
    const bool diag = (kt == qt);
#pragma unroll
    for (int t = 0; t < 4; ++t)
#pragma unroll
      for (int r = 0; r < 8; ++r) {
        float s = sacc[t][r] * 0.125f;  // 1/sqrt(64)
        if (diag) {
          int qrow = qBase + w * 16 + r + 8 * hf;
          int krow = kBase + t * 16 + ln;
          if (krow > qrow) s = -1e9f;
        }
        sacc[t][r] = s;
      }
    // online softmax: row stats live per-VGPR r (row m = r + 8*hf),
    // reduced across the 16 lanes holding that row.
    float sf[8];
#pragma unroll
    for (int r = 0; r < 8; ++r) {
      float mx = fmaxf(fmaxf(sacc[0][r], sacc[1][r]),
                       fmaxf(sacc[2][r], sacc[3][r]));
#pragma unroll
      for (int msk = 1; msk < 16; msk <<= 1)
        mx = fmaxf(mx, __shfl_xor(mx, msk, 32));
      float mnew = fmaxf(mrow[r], mx);
      sf[r] = __expf(mrow[r] - mnew);
      mrow[r] = mnew;
      float sum = 0.f;
#pragma unroll
      for (int t = 0; t < 4; ++t) {
        float p = __expf(sacc[t][r] - mnew);
        sacc[t][r] = p;
        sum += p;
      }
#pragma unroll
      for (int msk = 1; msk < 16; msk <<= 1)
        sum += __shfl_xor(sum, msk, 32);
      lrow[r] = lrow[r] * sf[r] + sum;
    }
    // rescale O, spill P (D-layout) to wave-private LDS in A-layout order
#pragma unroll
    for (int t = 0; t < 4; ++t)
#pragma unroll
      for (int r = 0; r < 8; ++r) {
        co[t][r] *= sf[r];
        Pw[(r + 8 * hf) * 72 + t * 16 + ln] = f2bf(sacc[t][r]);
      }
    // O += P V : contraction over 64 keys, two K=32 steps
#pragma unroll
    for (int kk = 0; kk < 64; kk += 32) {
      v16bf af = load_frag(&Pw[ln * 72 + kk], hf);
#pragma unroll
      for (int t = 0; t < 4; ++t) {
        v16bf bfg = load_frag(&Vt[(t * 16 + ln) * 72 + kk], hf);
        co[t] = __builtin_amdgcn_wmma_f32_16x16x32_bf16(
            false, af, false, bfg, (short)0, co[t], false, false);
      }
    }
  }
  // normalize and store to [B, S, H*DV] bf16 for the output projection
#pragma unroll
  for (int r = 0; r < 8; ++r) {
    float inv = 1.0f / lrow[r];
    int row = qBase + w * 16 + r + 8 * hf;
    size_t rowOff = ((size_t)(bq * Sseq + row)) * (Hh * Dk);
#pragma unroll
    for (int t = 0; t < 4; ++t) {
      int col = h * Dk + t * 16 + ln;
      attn[rowOff + col] = f2bf(co[t][r] * inv);
    }
  }
}

extern "C" void kernel_launch(void* const* d_in, const int* in_sizes, int n_in,
                              void* d_out, int out_size, void* d_ws, size_t ws_size,
                              hipStream_t stream) {
  (void)in_sizes; (void)n_in; (void)out_size; (void)ws_size;
  const float* Q  = (const float*)d_in[0];
  const float* K  = (const float*)d_in[1];
  const float* V  = (const float*)d_in[2];
  const float* Wq = (const float*)d_in[3];
  const float* Wk = (const float*)d_in[4];
  const float* Wv = (const float*)d_in[5];
  const float* Wo = (const float*)d_in[6];

  char* ws = (char*)d_ws;
  size_t off = 0;
  auto alloc = [&](size_t bytes) {
    void* p = ws + off;
    off += (bytes + 255) & ~(size_t)255;
    return p;
  };
  const size_t actBytes = (size_t)Mrows * Dmodel * sizeof(bh);  // 8 MB each
  bh* qhd   = (bh*)alloc(actBytes);  // head-major projections [B*H][S][64]
  bh* khd   = (bh*)alloc(actBytes);
  bh* vhd   = (bh*)alloc(actBytes);
  bh* attnb = (bh*)alloc(actBytes);  // [B,S,H*DV]

  dim3 gg(Dmodel / 64, Mrows / 128);  // (16, 32)
  k_gemm<0, float><<<gg, 256, 0, stream>>>(Q, Wq, qhd, Dmodel, Dmodel);
  k_gemm<0, float><<<gg, 256, 0, stream>>>(K, Wk, khd, Dmodel, Dmodel);
  k_gemm<0, float><<<gg, 256, 0, stream>>>(V, Wv, vhd, Dmodel, Dmodel);

  dim3 gf(Sseq / 64, Bsz * Hh);       // (32, 32)
  k_flash<<<gf, 128, FLASH_SMEM, stream>>>(qhd, khd, vhd, attnb);

  k_gemm<1, bh><<<gg, 256, 0, stream>>>(attnb, Wo, d_out, Dmodel, Dmodel);
}
